// MultiHeadAttention_20469814133179
// MI455X (gfx1250) — matrix operand order repaired and checked by reading, not compiled, not run on hardware
//
#include <hip/hip_runtime.h>
#include <hip/hip_bf16.h>

// ---------------------------------------------------------------------------
// Problem constants (from the reference)
// ---------------------------------------------------------------------------
#define DM    256          // D_MODEL
#define NHD   4            // NUM_HEADS
#define DKD   64           // D_K
#define NG    4096         // N_GROUPS
#define SEN   8192         // SEN_LEN
#define PLEN  64           // PATH_LEN
#define OUT_ATT_OFF (SEN * DM)   // att appended after `returned`

typedef __attribute__((ext_vector_type(16))) _Float16 v16h;
typedef __attribute__((ext_vector_type(8)))  _Float16 v8h;
typedef __attribute__((ext_vector_type(8)))  float    v8f;

// ---------------------------------------------------------------------------
// Kernel 1: convert W_k / W_v / W_q / W_out (256x256 f32 [out][in]) to f16
// ---------------------------------------------------------------------------
__global__ void prep_weights_kernel(const float* __restrict__ Wk,
                                    const float* __restrict__ Wv,
                                    const float* __restrict__ Wq,
                                    const float* __restrict__ Wo,
                                    _Float16* __restrict__ wk16,
                                    _Float16* __restrict__ wv16,
                                    _Float16* __restrict__ wq16,
                                    _Float16* __restrict__ wo16) {
  int idx = blockIdx.x * blockDim.x + threadIdx.x;   // 65536 threads total
  wk16[idx] = (_Float16)Wk[idx];
  wv16[idx] = (_Float16)Wv[idx];
  wq16[idx] = (_Float16)Wq[idx];
  wo16[idx] = (_Float16)Wo[idx];
}

// ---------------------------------------------------------------------------
// Kernel 2: exclusive prefix-sum of edge_len -> per-group start offsets
// ---------------------------------------------------------------------------
__global__ void scan_kernel(const int* __restrict__ edge_len,
                            int* __restrict__ starts) {
  if (threadIdx.x == 0 && blockIdx.x == 0) {
    int acc = 0;
    for (int g = 0; g < NG; ++g) { starts[g] = acc; acc += edge_len[g]; }
  }
}

// ---------------------------------------------------------------------------
// Kernel 3: gather q rows at token_ids and convert to f16
// ---------------------------------------------------------------------------
__global__ __launch_bounds__(256)
void gather_q_kernel(const float* __restrict__ q,
                     const int* __restrict__ token_ids,
                     _Float16* __restrict__ qg16) {
  const int g = blockIdx.x;
  const int tid = threadIdx.x;
  qg16[(size_t)g * DM + tid] = (_Float16)q[(size_t)token_ids[g] * DM + tid];
}

// ---------------------------------------------------------------------------
// Kernel 4: path attention (tiny): pair_q, att (output #2), path_result
// ---------------------------------------------------------------------------
__global__ __launch_bounds__(256)
void path_kernel(const float* __restrict__ path,
                 const float* __restrict__ q,
                 const float* __restrict__ W_qpair,
                 const float* __restrict__ W_pk,
                 const float* __restrict__ W_pv,
                 const float* __restrict__ W_out,
                 const float* __restrict__ b_out,
                 const int*   __restrict__ pair,
                 const int*   __restrict__ rev,
                 float* __restrict__ att_out,     // 256 = H*PLEN
                 float* __restrict__ pathres) {   // 256
  __shared__ float s_pairq[DM];
  __shared__ float s_att[NHD][PLEN];
  __shared__ float s_pv[DM];
  __shared__ float s_m[NHD];
  __shared__ float s_s[NHD];

  const int tid = threadIdx.x;
  const int rv  = rev[0];
  const int src = (rv == 0) ? pair[0] : pair[1];
  const int tar = (rv == 0) ? pair[1] : pair[0];

  // pair_q[i] = dot(concat(q[src], q[tar]), W_qpair[i, :])
  {
    float acc = 0.f;
    const float* wr = W_qpair + (size_t)tid * (2 * DM);
    const float* qs = q + (size_t)src * DM;
    const float* qt = q + (size_t)tar * DM;
    for (int j = 0; j < DM; ++j) acc += qs[j] * wr[j];
    for (int j = 0; j < DM; ++j) acc += qt[j] * wr[DM + j];
    s_pairq[tid] = acc;
  }
  __syncthreads();

  const int h = tid >> 6;       // 0..3
  const int p = tid & 63;       // 0..63

  // logits[h][p] = (1/8) * sum_d pair_q[h*64+d] * dot(path[p], W_pk[h*64+d])
  {
    float lg = 0.f;
    const float* prow = path + (size_t)p * DM;
    for (int d = 0; d < DKD; ++d) {
      const float* wr = W_pk + (size_t)(h * DKD + d) * DM;
      float pk = 0.f;
      for (int j = 0; j < DM; ++j) pk += prow[j] * wr[j];
      lg += s_pairq[h * DKD + d] * pk;
    }
    s_att[h][p] = lg * 0.125f;
  }
  __syncthreads();

  if (tid < NHD) {
    float m = -3.4e38f;
    for (int pp = 0; pp < PLEN; ++pp) m = fmaxf(m, s_att[tid][pp]);
    float s = 0.f;
    for (int pp = 0; pp < PLEN; ++pp) s += __expf(s_att[tid][pp] - m);
    s_m[tid] = m; s_s[tid] = s;
  }
  __syncthreads();
  s_att[h][p] = __expf(s_att[h][p] - s_m[h]) / s_s[h];
  __syncthreads();

  att_out[tid] = s_att[h][p];   // (H,1,PLEN) row-major = h*64+p

  // path_vals[h][d] = sum_p att[h][p] * dot(path[p], W_pv[h*64+d])
  {
    const int d = tid & 63;
    const float* wr = W_pv + (size_t)(h * DKD + d) * DM;
    float acc = 0.f;
    for (int pp = 0; pp < PLEN; ++pp) {
      const float* prow = path + (size_t)pp * DM;
      float pv = 0.f;
      for (int j = 0; j < DM; ++j) pv += prow[j] * wr[j];
      acc += s_att[h][pp] * pv;
    }
    s_pv[h * DKD + d] = acc;
  }
  __syncthreads();

  // path_result[i] = dot(path_vals, W_out[i,:]) + b_out[i]
  {
    float acc = b_out[tid];
    const float* wr = W_out + (size_t)tid * DM;
    for (int j = 0; j < DM; ++j) acc += s_pv[j] * wr[j];
    pathres[tid] = acc;
  }
}

// ---------------------------------------------------------------------------
// Kernel 5: generic batched GEMM  C[M x 256] = A[M x 256](f16) @ W(f16)^T
// W is [out][in] row-major (so C[m][o] = dot(A[m,:], W[o,:])).
// One block per 16-row M-tile; 8 waves x 2 N-tiles; K-loop of 8 x 32.
// WMMA operand layouts per CDNA5 ISA 7.12.2 (see tile_project_wmma).
// ---------------------------------------------------------------------------
__global__ __launch_bounds__(256)
void gemm_aw_kernel(const _Float16* __restrict__ A,
                    const _Float16* __restrict__ W,
                    float* __restrict__ C) {
  const int mt   = blockIdx.x;
  const int tid  = threadIdx.x;
  const int lane = tid & 31;
  const int wave = tid >> 5;
  const int m    = lane & 15;   // A row within tile / B column
  const int hi   = lane >> 4;

  const _Float16* arow = A + (size_t)(mt * 16 + m) * DM;

  #pragma unroll
  for (int s = 0; s < 2; ++s) {
    const int ntile = wave * 2 + s;
    v8f c = {};
    #pragma unroll
    for (int kc = 0; kc < 8; ++kc) {
      v8h alo = *(const v8h*)(arow + kc * 32 + hi * 8);
      v8h ahi = *(const v8h*)(arow + kc * 32 + 16 + hi * 8);
      const _Float16* bbase =
          W + (size_t)(ntile * 16 + m) * DM + kc * 32 + hi * 8;
      v8h blo = *(const v8h*)(bbase);
      v8h bhi = *(const v8h*)(bbase + 16);
      v16h av, bv;
      #pragma unroll
      for (int j = 0; j < 8; ++j) {
        av[j] = alo[j]; av[j + 8] = ahi[j];
        bv[j] = blo[j]; bv[j + 8] = bhi[j];
      }
      c = __builtin_amdgcn_wmma_f32_16x16x32_f16(
              false, av, false, bv, (short)0, c, false, false);
    }
    #pragma unroll
    for (int r = 0; r < 8; ++r)
      C[(size_t)(mt * 16 + hi * 8 + r) * DM + ntile * 16 + m] = c[r];
  }
}

// ---------------------------------------------------------------------------
// Shared WMMA tile projection: 16 edge rows (f32 in global) x W^T (f16)
// -> 16x256 f32 tile in LDS. 8 waves, 2 N-tiles of 16 outputs each.
// ---------------------------------------------------------------------------
__device__ __forceinline__ void tile_project_wmma(
    const float* __restrict__ rows,          // k or v (E x 256 f32)
    const _Float16* __restrict__ w16,        // 256x256 f16, row-major [out][in]
    int start, int t, int len,
    _Float16 (*s_kv)[264], float (*s_proj)[260],
    int tid) {
  // stage 16 rows x 256 cols of f32 -> f16 LDS (zero-pad past len)
  {
    const int r  = tid >> 4;            // 0..15 edge row
    const int c0 = (tid & 15) << 4;     // 0,16,...,240
    const int e  = t * 16 + r;
    if (e < len) {
      const float* src = rows + (size_t)(start + e) * DM + c0;
      if (e + 16 < len)
        __builtin_prefetch(rows + (size_t)(start + e + 16) * DM + c0, 0, 0);
      #pragma unroll
      for (int j = 0; j < 16; ++j) s_kv[r][c0 + j] = (_Float16)src[j];
    } else {
      #pragma unroll
      for (int j = 0; j < 16; ++j) s_kv[r][c0 + j] = (_Float16)0.0f;
    }
  }
  __syncthreads();

  const int lane = tid & 31;
  const int wave = tid >> 5;
  const int m    = lane & 15;
  const int hi   = lane >> 4;

  #pragma unroll
  for (int s = 0; s < 2; ++s) {
    const int ntile = wave * 2 + s;     // 16 N-tiles over 8 waves
    v8f c = {};
    #pragma unroll
    for (int kc = 0; kc < 8; ++kc) {
      const _Float16* abase = &s_kv[m][kc * 32];
      v8h alo = *(const v8h*)(abase + hi * 8);        // K = kc*32 + hi*8 + 0..7
      v8h ahi = *(const v8h*)(abase + 16 + hi * 8);   // K = kc*32 + 16 + hi*8 + 0..7
      const _Float16* bbase =
          w16 + (size_t)(ntile * 16 + m) * DM + kc * 32 + hi * 8;
      v8h blo = *(const v8h*)(bbase);                 // K = kc*32 + hi*8 + 0..7
      v8h bhi = *(const v8h*)(bbase + 16);            // K = kc*32 + 16 + hi*8 + 0..7
      v16h av, bv;
      #pragma unroll
      for (int j = 0; j < 8; ++j) {
        av[j] = alo[j]; av[j + 8] = ahi[j];
        bv[j] = blo[j]; bv[j + 8] = bhi[j];
      }
      c = __builtin_amdgcn_wmma_f32_16x16x32_f16(
              false, av, false, bv, (short)0, c, false, false);
    }
    #pragma unroll
    for (int r2 = 0; r2 < 8; ++r2)
      s_proj[hi * 8 + r2][ntile * 16 + m] = c[r2];
  }
  __syncthreads();
}

// ---------------------------------------------------------------------------
// Kernel 6: fused per-group attention core. One block per group, 256 threads.
// Reads precomputed qsel (f32); writes hid (f16) for the batched W_out GEMM.
// ---------------------------------------------------------------------------
__global__ __launch_bounds__(256)
void group_attn_kernel(const float* __restrict__ k,
                       const float* __restrict__ v,
                       const int*   __restrict__ edge_len,
                       const _Float16* __restrict__ wk16,
                       const _Float16* __restrict__ wv16,
                       const int*   __restrict__ starts,
                       const float* __restrict__ qsel,     // NG x 256 f32
                       _Float16*    __restrict__ hid16) {  // NG x 256 f16
  __shared__ _Float16 s_kv[16][264];    // 16-edge f16 staging tile (padded)
  __shared__ float    s_proj[16][260];  // projected 16x256 tile (padded)
  __shared__ float    s_p[128][16];     // logits -> softmax p (len<=126)
  __shared__ float    s_qs[DM];         // projected group query (H*dk)
  __shared__ float    s_hid[DM];
  __shared__ float    s_lm[16];
  __shared__ float    s_den[16];

  const int g    = blockIdx.x;
  const int tid  = threadIdx.x;
  const int len  = edge_len[g];
  const int start= starts[g];

  s_qs[tid]  = qsel[(size_t)g * DM + tid];
  s_hid[tid] = 0.f;
  __syncthreads();

  const int nt = (len + 15) >> 4;

  // --- pass 1: k-projection tiles + cross-head logits -----------------------
  for (int t = 0; t < nt; ++t) {
    tile_project_wmma(k, wk16, start, t, len, s_kv, s_proj, tid);
    {
      const int el = tid >> 4;            // edge in tile
      const int a  = (tid >> 2) & 3;      // query head
      const int b  = tid & 3;             // key head
      const int e  = t * 16 + el;
      if (e < len) {
        const float* qa = &s_qs[a * DKD];
        const float* kb = &s_proj[el][b * DKD];
        float acc = 0.f;
        #pragma unroll 8
        for (int d = 0; d < DKD; ++d) acc += qa[d] * kb[d];
        s_p[e][a * 4 + b] = acc * 0.125f;
      }
    }
    __syncthreads();
  }

  // --- segment softmax over the group's edges -------------------------------
  if (tid < 16) {
    float m = -3.4e38f;
    for (int e = 0; e < len; ++e) m = fmaxf(m, s_p[e][tid]);
    float s = 0.f;
    for (int e = 0; e < len; ++e) s += __expf(s_p[e][tid] - m);
    s_lm[tid] = m; s_den[tid] = s;
  }
  __syncthreads();
  for (int idx = tid; idx < len * 16; idx += 256) {
    const int e = idx >> 4, ab = idx & 15;
    s_p[e][ab] = __expf(s_p[e][ab] - s_lm[ab]) / s_den[ab];
  }
  __syncthreads();

  // --- pass 2: v-projection tiles + weighted accumulate ---------------------
  for (int t = 0; t < nt; ++t) {
    tile_project_wmma(v, wv16, start, t, len, s_kv, s_proj, tid);
    {
      const int a = tid >> 6;             // output head
      const int d = tid & 63;             // dim
      int lim = len - t * 16; if (lim > 16) lim = 16;
      float acc = 0.f;
      for (int el = 0; el < lim; ++el) {
        const float* pe = s_p[t * 16 + el];
        const float* ve = s_proj[el];
        #pragma unroll
        for (int b = 0; b < 4; ++b)
          acc += pe[a * 4 + b] * ve[b * DKD + d];
      }
      s_hid[tid] += acc;                  // each thread owns one element
    }
    __syncthreads();
  }

  hid16[(size_t)g * DM + tid] = (_Float16)s_hid[tid];
}

// ---------------------------------------------------------------------------
// Kernel 7: epilogue. nh = gemm_out + b_out; W_agg mix for src/tar groups;
// scatter into out[token_ids[g]].
// ---------------------------------------------------------------------------
__global__ __launch_bounds__(256)
void epilogue_kernel(const float* __restrict__ nhraw,     // NG x 256
                     const float* __restrict__ b_out,
                     const float* __restrict__ W_agg,
                     const float* __restrict__ pathres,
                     const int*   __restrict__ token_ids,
                     const int*   __restrict__ pair,
                     const int*   __restrict__ rev,
                     float* __restrict__ out) {
  __shared__ float s_nh[DM];
  const int g   = blockIdx.x;
  const int tid = threadIdx.x;
  const int tok = token_ids[g];

  s_nh[tid] = nhraw[(size_t)g * DM + tid] + b_out[tid];
  __syncthreads();

  const int rv  = rev[0];
  const int src = (rv == 0) ? pair[0] : pair[1];
  const int tar = (rv == 0) ? pair[1] : pair[0];

  float val;
  if (tok == src) {            // feat_path = W_agg @ concat(new_hid, pr)
    const float* wr = W_agg + (size_t)tid * (2 * DM);
    float acc = 0.f;
    for (int j = 0; j < DM; ++j) acc += s_nh[j]    * wr[j];
    for (int j = 0; j < DM; ++j) acc += pathres[j] * wr[DM + j];
    val = acc;
  } else if (tok == tar) {     // path_feat = W_agg @ concat(pr, new_hid)
    const float* wr = W_agg + (size_t)tid * (2 * DM);
    float acc = 0.f;
    for (int j = 0; j < DM; ++j) acc += pathres[j] * wr[j];
    for (int j = 0; j < DM; ++j) acc += s_nh[j]    * wr[DM + j];
    val = acc;
  } else {
    val = s_nh[tid];
  }
  out[(size_t)tok * DM + tid] = val;   // token_ids are unique -> no conflicts
}

// ---------------------------------------------------------------------------
// Host launcher
// ---------------------------------------------------------------------------
extern "C" void kernel_launch(void* const* d_in, const int* in_sizes, int n_in,
                              void* d_out, int out_size, void* d_ws, size_t ws_size,
                              hipStream_t stream) {
  const float* path    = (const float*)d_in[0];
  const float* q       = (const float*)d_in[2];
  const float* k       = (const float*)d_in[3];
  const float* v       = (const float*)d_in[4];
  const int*   edge_len= (const int*)d_in[6];
  const int*   token_ids=(const int*)d_in[7];
  const int*   pair    = (const int*)d_in[8];
  const int*   rev     = (const int*)d_in[9];
  const float* W_qpair = (const float*)d_in[10];
  const float* W_pk    = (const float*)d_in[11];
  const float* W_pv    = (const float*)d_in[12];
  const float* W_q     = (const float*)d_in[13];
  const float* W_k     = (const float*)d_in[14];
  const float* W_v     = (const float*)d_in[15];
  const float* W_out   = (const float*)d_in[16];
  const float* b_out   = (const float*)d_in[17];
  const float* W_agg   = (const float*)d_in[18];
  float* out = (float*)d_out;

  char* ws = (char*)d_ws;
  _Float16* wk16   = (_Float16*)(ws);                       // 128 KB
  _Float16* wv16   = (_Float16*)(ws + (1 << 17));           // 128 KB
  _Float16* wq16   = (_Float16*)(ws + 2 * (1 << 17));       // 128 KB
  _Float16* wo16   = (_Float16*)(ws + 3 * (1 << 17));       // 128 KB
  int*      starts = (int*)(ws + 4 * (1 << 17));            // 16 KB
  float*    pathres= (float*)(ws + 4 * (1 << 17) + 16384);  // 1 KB
  char*     base2  = ws + 4 * (1 << 17) + 16384 + 1024;
  _Float16* qg16   = (_Float16*)(base2);                    // 2 MB (NG x 256 f16)
  float*    qsel   = (float*)(base2 + 2097152);             // 4 MB (NG x 256 f32)
  _Float16* hid16  = (_Float16*)(base2 + 2097152 + 4194304);// 2 MB
  float*    nhraw  = (float*)(base2 + 2097152 + 4194304 + 2097152); // 4 MB

  prep_weights_kernel<<<256, 256, 0, stream>>>(W_k, W_v, W_q, W_out,
                                               wk16, wv16, wq16, wo16);
  scan_kernel<<<1, 32, 0, stream>>>(edge_len, starts);
  gather_q_kernel<<<NG, 256, 0, stream>>>(q, token_ids, qg16);
  path_kernel<<<1, 256, 0, stream>>>(path, q, W_qpair, W_pk, W_pv, W_out, b_out,
                                     pair, rev, out + OUT_ATT_OFF, pathres);
  // qsel = qg16 @ W_q^T  (WMMA GEMM, 256 M-tiles)
  gemm_aw_kernel<<<NG / 16, 256, 0, stream>>>(qg16, wq16, qsel);
  // returned = q.at[token_ids].set(vals): bulk copy q, then scatter rows.
  hipMemcpyAsync(out, q, (size_t)SEN * DM * sizeof(float),
                 hipMemcpyDeviceToDevice, stream);
  // fused attention core -> hid16
  group_attn_kernel<<<NG, 256, 0, stream>>>(k, v, edge_len, wk16, wv16,
                                            starts, qsel, hid16);
  // nh_raw = hid16 @ W_out^T  (WMMA GEMM)
  gemm_aw_kernel<<<NG / 16, 256, 0, stream>>>(hid16, wo16, nhraw);
  epilogue_kernel<<<NG, 256, 0, stream>>>(nhraw, b_out, W_agg, pathres,
                                          token_ids, pair, rev, out);
}
